// MultiHeadSelfAttention_78331613545069
// MI455X (gfx1250) — matrix operand order repaired and checked by reading, not compiled, not run on hardware
//
#include <hip/hip_runtime.h>
#include <hip/hip_bf16.h>

// Problem constants (from reference)
#define L_    2048
#define NB    2
#define E_    1024
#define H_    16
#define DHD   64
#define MROWS (L_ * NB)     // 4096 flattened (l,n) rows
#define F3    (3 * E_)      // 3072 qkv output cols
#define SCALE 0.125f        // DH^-0.5

typedef __bf16 bf16;
typedef __attribute__((ext_vector_type(8)))  bf16  bf16x8;
typedef __attribute__((ext_vector_type(16))) bf16  bf16x16;
typedef __attribute__((ext_vector_type(8)))  float floatx8;

static __device__ inline bf16 f2bf(float f) {
  union { float f; unsigned u; } in; in.f = f;
  unsigned u = in.u;
  unsigned r = u + 0x7FFFu + ((u >> 16) & 1u);   // round-to-nearest-even
  union { unsigned short s; bf16 b; } out; out.s = (unsigned short)(r >> 16);
  return out.b;
}

// A-fragment 16x32 bf16 (M x K), memory row-major, K contiguous, row stride ld.
// Per ISA: lanes 0-15 row=lane, K {0..7,16..23}; lanes 16-31 row=lane-16, K {8..15,24..31}.
static __device__ inline bf16x16 load_a_frag(const bf16* __restrict__ base, int ld, int lane) {
  int r    = lane & 15;
  int half = lane >> 4;
  const bf16* p = base + (size_t)r * ld + half * 8;
  bf16x8 c0 = *(const bf16x8*)(p);
  bf16x8 c1 = *(const bf16x8*)(p + 16);
  bf16x16 a;
#pragma unroll
  for (int i = 0; i < 8; ++i) { a[i] = c0[i]; a[i + 8] = c1[i]; }
  return a;
}

// B-fragment 32x16 bf16 (K x N), sourced from B^T rows (N rows, K contiguous), row stride ld.
// Per ISA: lanes 0-15 col=lane K {0..7,16..23}; lanes 16-31 col=lane-16 K {8..15,24..31}.
static __device__ inline bf16x16 load_b_frag(const bf16* __restrict__ base, int ld, int lane) {
  int c    = lane & 15;
  int half = lane >> 4;
  const bf16* p = base + (size_t)c * ld + half * 8;
  bf16x8 c0 = *(const bf16x8*)(p);
  bf16x8 c1 = *(const bf16x8*)(p + 16);
  bf16x16 b;
#pragma unroll
  for (int i = 0; i < 8; ++i) { b[i] = c0[i]; b[i + 8] = c1[i]; }
  return b;
}

static __device__ inline floatx8 wmma_bf16(bf16x16 a, bf16x16 b, floatx8 c) {
  return __builtin_amdgcn_wmma_f32_16x16x32_bf16(false, a, false, b, (short)0, c, false, false);
}

// ---------------------------------------------------------------- kernel 1
__global__ void convert_bf16_kernel(const float* __restrict__ x,
                                    const float* __restrict__ wqkv,
                                    const float* __restrict__ wout,
                                    bf16* __restrict__ xb,
                                    bf16* __restrict__ wqkvb,
                                    bf16* __restrict__ woutb) {
  const long nx = (long)MROWS * E_;
  const long nq = (long)F3 * E_;
  const long nw = (long)E_ * E_;
  const long total = nx + nq + nw;
  long stride = (long)gridDim.x * blockDim.x;
  for (long t = (long)blockIdx.x * blockDim.x + threadIdx.x; t < total; t += stride) {
    if (t < nx)            xb[t]              = f2bf(x[t]);
    else if (t < nx + nq)  wqkvb[t - nx]      = f2bf(wqkv[t - nx]);
    else                   woutb[t - nx - nq] = f2bf(wout[t - nx - nq]);
  }
}

// ---------------------------------------------------------------- kernel 2
// C[m,f] = sum_k X[m,k] * Wqkv[f,k]; scatter into per-head q (scaled), k, vT.
// Block 256 thr = 8 waves (4 M x 2 N), wave tile 32x64 -> block tile 128x128.
__global__ __launch_bounds__(256) void qkv_gemm_kernel(const bf16* __restrict__ A,
                                                       const bf16* __restrict__ B,
                                                       bf16* __restrict__ qh,
                                                       bf16* __restrict__ kh,
                                                       bf16* __restrict__ vT) {
  int lane = threadIdx.x & 31;
  int wave = threadIdx.x >> 5;
  int wm = wave & 3, wn = wave >> 2;
  int m0 = blockIdx.x * 128 + wm * 32;
  int n0 = blockIdx.y * 128 + wn * 64;

  floatx8 c[2][4] = {};
  for (int k0 = 0; k0 < E_; k0 += 32) {
    __builtin_prefetch(A + (size_t)m0 * E_ + k0 + 64, 0, 1);
    bf16x16 a0 = load_a_frag(A + (size_t)m0 * E_ + k0, E_, lane);
    bf16x16 a1 = load_a_frag(A + (size_t)(m0 + 16) * E_ + k0, E_, lane);
    bf16x16 b0 = load_b_frag(B + (size_t)(n0 +  0) * E_ + k0, E_, lane);
    bf16x16 b1 = load_b_frag(B + (size_t)(n0 + 16) * E_ + k0, E_, lane);
    bf16x16 b2 = load_b_frag(B + (size_t)(n0 + 32) * E_ + k0, E_, lane);
    bf16x16 b3 = load_b_frag(B + (size_t)(n0 + 48) * E_ + k0, E_, lane);
    c[0][0] = wmma_bf16(a0, b0, c[0][0]);
    c[0][1] = wmma_bf16(a0, b1, c[0][1]);
    c[0][2] = wmma_bf16(a0, b2, c[0][2]);
    c[0][3] = wmma_bf16(a0, b3, c[0][3]);
    c[1][0] = wmma_bf16(a1, b0, c[1][0]);
    c[1][1] = wmma_bf16(a1, b1, c[1][1]);
    c[1][2] = wmma_bf16(a1, b2, c[1][2]);
    c[1][3] = wmma_bf16(a1, b3, c[1][3]);
  }

  // C-tile layout: N = lane&15; M = vgpr + 8*(lane>=16).
  int ncol = lane & 15;
  int mb   = (lane >> 4) * 8;
#pragma unroll
  for (int mt = 0; mt < 2; ++mt)
#pragma unroll
    for (int nt = 0; nt < 4; ++nt)
#pragma unroll
      for (int r = 0; r < 8; ++r) {
        int m = m0 + mt * 16 + mb + r;
        int f = n0 + nt * 16 + ncol;
        float val = c[mt][nt][r];
        int l = m >> 1, n = m & 1;           // m = l*NB + n, NB=2
        int which = f >> 10;                 // 0=q 1=k 2=v
        int e  = f & (E_ - 1);
        int h  = e >> 6, dh = e & (DHD - 1);
        int bh = n * H_ + h;
        if (which == 0)      qh[((size_t)bh * L_ + l) * DHD + dh] = f2bf(val * SCALE);
        else if (which == 1) kh[((size_t)bh * L_ + l) * DHD + dh] = f2bf(val);
        else                 vT[((size_t)bh * DHD + dh) * L_ + l] = f2bf(val);
      }
}

// ---------------------------------------------------------------- kernel 3
// Flash attention: one wave per 32-row query tile (2 x 16-row subtiles);
// 4 waves/block. K/V fragments are loaded once per 32-key step and reused
// by both query subtiles (halves K/V global traffic per WMMA).
__global__ __launch_bounds__(128) void attn_kernel(const bf16* __restrict__ qh,
                                                   const bf16* __restrict__ kh,
                                                   const bf16* __restrict__ vT,
                                                   bf16* __restrict__ aout) {
  __shared__ bf16 ldsP[4][32][32];   // per-wave P staging (C-layout -> A-layout)

  int lane = threadIdx.x & 31;
  int wave = threadIdx.x >> 5;
  int bh = blockIdx.x;                       // head-batch 0..31  (bh = n*H + h)
  int qtile = blockIdx.y * 4 + wave;         // 0..63
  int q0 = qtile * 32;

  const bf16* qb = qh + (size_t)bh * L_ * DHD;
  const bf16* kb = kh + (size_t)bh * L_ * DHD;
  const bf16* vb = vT + (size_t)bh * DHD * L_;

  // Q fragments: 2 row-subtiles x 2 K-dim(DH) chunks — loop invariant.
  bf16x16 qa[2][2];
#pragma unroll
  for (int u = 0; u < 2; ++u) {
    qa[u][0] = load_a_frag(qb + (size_t)(q0 + u * 16) * DHD + 0,  DHD, lane);
    qa[u][1] = load_a_frag(qb + (size_t)(q0 + u * 16) * DHD + 32, DHD, lane);
  }

  floatx8 o[2][4] = {};
  float mrow[2][8], lrow[2][8];
#pragma unroll
  for (int u = 0; u < 2; ++u)
#pragma unroll
    for (int r = 0; r < 8; ++r) { mrow[u][r] = -__builtin_inff(); lrow[u][r] = 0.f; }

  int ncol = lane & 15;
  int mb   = (lane >> 4) * 8;
  const float NEG = -__builtin_inff();

  for (int m0 = 0; m0 < q0 + 32; m0 += 32) {
    // ---- K fragments for this 32-key tile (shared by both q subtiles) ----
    bf16x16 kf[2][2];
#pragma unroll
    for (int t = 0; t < 2; ++t) {
      kf[t][0] = load_b_frag(kb + (size_t)(m0 + t * 16) * DHD + 0,  DHD, lane);
      kf[t][1] = load_b_frag(kb + (size_t)(m0 + t * 16) * DHD + 32, DHD, lane);
    }
    // ---- S = Q K^T : 2 q-subtiles x 2 key-subtiles ----
    floatx8 s[2][2];
#pragma unroll
    for (int u = 0; u < 2; ++u)
#pragma unroll
      for (int t = 0; t < 2; ++t) {
        floatx8 z = {};
        z = wmma_bf16(qa[u][0], kf[t][0], z);
        z = wmma_bf16(qa[u][1], kf[t][1], z);
        s[u][t] = z;
      }
    // ---- causal mask ----
#pragma unroll
    for (int u = 0; u < 2; ++u)
#pragma unroll
      for (int t = 0; t < 2; ++t)
#pragma unroll
        for (int r = 0; r < 8; ++r) {
          int kg = m0 + t * 16 + ncol;
          int qg = q0 + u * 16 + mb + r;
          if (kg > qg) s[u][t][r] = NEG;
        }
    // ---- online softmax per q subtile ----
    float alpha[2][8];
#pragma unroll
    for (int u = 0; u < 2; ++u)
#pragma unroll
      for (int r = 0; r < 8; ++r) {
        float v = fmaxf(s[u][0][r], s[u][1][r]);
        v = fmaxf(v, __shfl_xor(v, 1, 32));
        v = fmaxf(v, __shfl_xor(v, 2, 32));
        v = fmaxf(v, __shfl_xor(v, 4, 32));
        v = fmaxf(v, __shfl_xor(v, 8, 32));   // stays within each 16-lane half
        float mn = fmaxf(mrow[u][r], v);
        alpha[u][r] = __expf(mrow[u][r] - mn);
        mrow[u][r] = mn;
        float p0 = __expf(s[u][0][r] - mn);
        float p1 = __expf(s[u][1][r] - mn);
        float rs = p0 + p1;
        rs += __shfl_xor(rs, 1, 32);
        rs += __shfl_xor(rs, 2, 32);
        rs += __shfl_xor(rs, 4, 32);
        rs += __shfl_xor(rs, 8, 32);
        lrow[u][r] = lrow[u][r] * alpha[u][r] + rs;
        ldsP[wave][u * 16 + mb + r][ncol]      = f2bf(p0);
        ldsP[wave][u * 16 + mb + r][16 + ncol] = f2bf(p1);
      }
#pragma unroll
    for (int u = 0; u < 2; ++u)
#pragma unroll
      for (int t2 = 0; t2 < 4; ++t2)
#pragma unroll
        for (int r = 0; r < 8; ++r) o[u][t2][r] *= alpha[u][r];

    // ---- V fragments (shared by both q subtiles) ----
    bf16x16 vf[4];
#pragma unroll
    for (int t2 = 0; t2 < 4; ++t2)
      vf[t2] = load_b_frag(vb + (size_t)(t2 * 16) * L_ + m0, L_, lane);

    // ---- P as A-fragments via LDS relayout, then O += P * V ----
    {
      int rr   = lane & 15;
      int half = lane >> 4;
#pragma unroll
      for (int u = 0; u < 2; ++u) {
        bf16x8 c0 = *(const bf16x8*)&ldsP[wave][u * 16 + rr][half * 8];
        bf16x8 c1 = *(const bf16x8*)&ldsP[wave][u * 16 + rr][half * 8 + 16];
        bf16x16 pa;
#pragma unroll
        for (int i = 0; i < 8; ++i) { pa[i] = c0[i]; pa[i + 8] = c1[i]; }
#pragma unroll
        for (int t2 = 0; t2 < 4; ++t2)
          o[u][t2] = wmma_bf16(pa, vf[t2], o[u][t2]);
      }
    }
  }

  // ---- finalize: O /= l, store [l, n, h*64 + dh] bf16 ----
  int n = bh >> 4, h = bh & 15;
#pragma unroll
  for (int u = 0; u < 2; ++u)
#pragma unroll
    for (int t2 = 0; t2 < 4; ++t2)
#pragma unroll
      for (int r = 0; r < 8; ++r) {
        int lq = q0 + u * 16 + mb + r;
        float val = o[u][t2][r] / lrow[u][r];
        aout[((size_t)lq * NB + n) * E_ + h * DHD + t2 * 16 + ncol] = f2bf(val);
      }
}

// ---------------------------------------------------------------- kernel 4
// out[m,f] = sum_k AO[m,k] * Wout[f,k]  (f32 output)
// Block 256 thr = 8 waves (4 M x 2 N), wave tile 32x64 -> block tile 128x128.
__global__ __launch_bounds__(256) void out_gemm_kernel(const bf16* __restrict__ A,
                                                       const bf16* __restrict__ B,
                                                       float* __restrict__ out) {
  int lane = threadIdx.x & 31;
  int wave = threadIdx.x >> 5;
  int wm = wave & 3, wn = wave >> 2;
  int m0 = blockIdx.x * 128 + wm * 32;
  int n0 = blockIdx.y * 128 + wn * 64;

  floatx8 c[2][4] = {};
  for (int k0 = 0; k0 < E_; k0 += 32) {
    __builtin_prefetch(A + (size_t)m0 * E_ + k0 + 64, 0, 1);
    bf16x16 a0 = load_a_frag(A + (size_t)m0 * E_ + k0, E_, lane);
    bf16x16 a1 = load_a_frag(A + (size_t)(m0 + 16) * E_ + k0, E_, lane);
    bf16x16 b0 = load_b_frag(B + (size_t)(n0 +  0) * E_ + k0, E_, lane);
    bf16x16 b1 = load_b_frag(B + (size_t)(n0 + 16) * E_ + k0, E_, lane);
    bf16x16 b2 = load_b_frag(B + (size_t)(n0 + 32) * E_ + k0, E_, lane);
    bf16x16 b3 = load_b_frag(B + (size_t)(n0 + 48) * E_ + k0, E_, lane);
    c[0][0] = wmma_bf16(a0, b0, c[0][0]);
    c[0][1] = wmma_bf16(a0, b1, c[0][1]);
    c[0][2] = wmma_bf16(a0, b2, c[0][2]);
    c[0][3] = wmma_bf16(a0, b3, c[0][3]);
    c[1][0] = wmma_bf16(a1, b0, c[1][0]);
    c[1][1] = wmma_bf16(a1, b1, c[1][1]);
    c[1][2] = wmma_bf16(a1, b2, c[1][2]);
    c[1][3] = wmma_bf16(a1, b3, c[1][3]);
  }

  int ncol = lane & 15;
  int mb   = (lane >> 4) * 8;
#pragma unroll
  for (int mt = 0; mt < 2; ++mt)
#pragma unroll
    for (int nt = 0; nt < 4; ++nt)
#pragma unroll
      for (int r = 0; r < 8; ++r) {
        int m = m0 + mt * 16 + mb + r;
        int f = n0 + nt * 16 + ncol;
        out[(size_t)m * E_ + f] = c[mt][nt][r];
      }
}

// ---------------------------------------------------------------- launch
extern "C" void kernel_launch(void* const* d_in, const int* in_sizes, int n_in,
                              void* d_out, int out_size, void* d_ws, size_t ws_size,
                              hipStream_t stream) {
  const float* x    = (const float*)d_in[0];   // [L, N, E]
  const float* wqkv = (const float*)d_in[1];   // [3E, E]
  const float* wout = (const float*)d_in[2];   // [E, E]
  float* out = (float*)d_out;                  // [L, N, E]

  char* ws = (char*)d_ws;
  const size_t MB = 1u << 20;
  bf16* xb    = (bf16*)(ws + 0 * MB);   // 8 MiB  x bf16 [4096,1024]
  bf16* wqkvb = (bf16*)(ws + 8 * MB);   // 6 MiB  w_qkv bf16 [3072,1024]
  bf16* woutb = (bf16*)(ws + 14 * MB);  // 2 MiB  w_out bf16 [1024,1024]
  bf16* qh    = (bf16*)(ws + 16 * MB);  // 8 MiB  q bf16 [32,2048,64] (pre-scaled)
  bf16* kh    = (bf16*)(ws + 24 * MB);  // 8 MiB  k bf16 [32,2048,64]
  bf16* vT    = (bf16*)(ws + 32 * MB);  // 8 MiB  v^T bf16 [32,64,2048]
  bf16* aob   = (bf16*)(ws + 40 * MB);  // 8 MiB  attn out bf16 [4096,1024]

  convert_bf16_kernel<<<512, 256, 0, stream>>>(x, wqkv, wout, xb, wqkvb, woutb);
  qkv_gemm_kernel<<<dim3(MROWS / 128, F3 / 128), 256, 0, stream>>>(xb, wqkvb, qh, kh, vT);
  attn_kernel<<<dim3(NB * H_, (L_ / 32) / 4), 128, 0, stream>>>(qh, kh, vT, aob);
  out_gemm_kernel<<<dim3(MROWS / 128, E_ / 128), 256, 0, stream>>>(aob, woutb, out);
}
